// CICDM_89515708384061
// MI455X (gfx1250) — matrix operand order repaired and checked by reading, not compiled, not run on hardware
//
#include <hip/hip_runtime.h>
#include <hip/hip_bf16.h>
#include <math.h>

typedef _Float16 f16;
typedef __attribute__((ext_vector_type(8)))  _Float16 v8h;
typedef __attribute__((ext_vector_type(16))) _Float16 v16h;
typedef __attribute__((ext_vector_type(8)))  float    v8f;

#define BN    4096
#define NOUT  1024
#define KN    30

// ---------------- f32 -> f16 weight conversion ----------------
__global__ void cvt_f32_to_f16(const float* __restrict__ in, f16* __restrict__ out, int n) {
    int i = blockIdx.x * blockDim.x + threadIdx.x;
    if (i < n) out[i] = (f16)in[i];
}

// ---------------- Choquet stage: C[b,n] in f16 ----------------
__device__ __forceinline__ float pickF(int o, float F0, float F1, float F2,
                                       float F3, float F4, float F5) {
    float r = 1.0f;                 // index 6 -> min(1,1) = 1
    r = (o == 0) ? F0 : r;
    r = (o == 1) ? F1 : r;
    r = (o == 2) ? F2 : r;
    r = (o == 3) ? F3 : r;
    r = (o == 4) ? F4 : r;
    r = (o == 5) ? F5 : r;
    return r;
}

__global__ void choquet_kernel(const int*   __restrict__ stu_id,
                               const float* __restrict__ emb,
                               const int*   __restrict__ q_idx,
                               const float* __restrict__ fm_vars,
                               f16*         __restrict__ Cout) {
    __shared__ float s[KN];
    const int b = blockIdx.y;
    const int n = blockIdx.x * blockDim.x + threadIdx.x;   // 4 blocks x 256 = 1024

    if (threadIdx.x < KN) {
        int   sid = stu_id[b];
        float e   = emb[(size_t)sid * KN + threadIdx.x];
        s[threadIdx.x] = 1.0f / (1.0f + expf(-e));         // sigmoid(stu emb)
    }
    __syncthreads();

    int q0 = q_idx[n * 3 + 0], q1 = q_idx[n * 3 + 1], q2 = q_idx[n * 3 + 2];
    float v0 = s[q0], v1 = s[q1], v2 = s[q2];
    int   i0 = 0, i1 = 1, i2 = 2;

    // stable descending sort of 3 (matches jnp.argsort(-x)): swap on strict <
    if (v0 < v1) { float t = v0; v0 = v1; v1 = t; int u = i0; i0 = i1; i1 = u; }
    if (v1 < v2) { float t = v1; v1 = v2; v2 = t; int u = i1; i1 = i2; i2 = u; }
    if (v0 < v1) { float t = v0; v0 = v1; v1 = t; int u = i0; i0 = i1; i1 = u; }

    float d0 = v0 - v1, d1 = v1 - v2, d2 = v2;
    int o0 = (1 << i0) - 1;                  // in {0,1,3}
    int o1 = (1 << i0) + (1 << i1) - 1;      // in {2,4,5}  (o2 == 6 always)

    float c0 = fabsf(fm_vars[0 * NOUT + n]);
    float c1 = fabsf(fm_vars[1 * NOUT + n]);
    float c2 = fabsf(fm_vars[2 * NOUT + n]);
    float c3 = fabsf(fm_vars[3 * NOUT + n]);
    float c4 = fabsf(fm_vars[4 * NOUT + n]);
    float c5 = fabsf(fm_vars[5 * NOUT + n]);
    float F0 = fminf(c0, 1.0f);
    float F1 = fminf(c1, 1.0f);
    float F2 = fminf(fmaxf(c0, c1) + c2, 1.0f);
    float F3 = fminf(c3, 1.0f);
    float F4 = fminf(fmaxf(c3, c0) + c4, 1.0f);
    float F5 = fminf(fmaxf(c3, c1) + c5, 1.0f);

    float Cv = d0 * pickF(o0, F0, F1, F2, F3, F4, F5)
             + d1 * pickF(o1, F0, F1, F2, F3, F4, F5)
             + d2;                            // FM[6] == 1

    Cout[(size_t)b * NOUT + n] = (f16)Cv;
}

// ---------------- WMMA GEMM: out[m,n] = act(sum_k A[m,k]*W[n,k] + bias[n]) ----
// A: (M x K) f16 row-major.  W: (N x K) f16 row-major (as given by reference).
// ACT 0: relu -> f16 out.  ACT 1: sigmoid -> f32 out.
// 8 waves / block, one 16x16 D tile per wave.
template <int ACT>
__global__ void gemm_wmma(const f16* __restrict__ A, const f16* __restrict__ W,
                          const float* __restrict__ bias, void* __restrict__ outv,
                          int M, int N, int K) {
    const int wave = threadIdx.x >> 5;
    const int lane = threadIdx.x & 31;
    const int h    = lane >> 4;     // lane half: 0 or 1
    const int r    = lane & 15;

    const int mTiles = M >> 4;
    int t = blockIdx.x * 8 + wave;  // grid sized exactly: t < mTiles*(N>>4)
    const int nT = t / mTiles;      // 8 consecutive waves share nT -> B frag reuse
    const int mT = t - nT * mTiles;

    const f16* aRow = A + (size_t)(mT * 16 + r) * K;   // A lane row = mT*16 + r
    const f16* wRow = W + (size_t)(nT * 16 + r) * K;   // B lane col = nT*16 + r

    const float bv = bias[nT * 16 + r];                // C/D: N = lane%16 fixed
    v8f acc = {bv, bv, bv, bv, bv, bv, bv, bv};

    for (int k0 = 0; k0 < K; k0 += 32) {
        // A 16x32 f16 frag: lane holds contiguous K = k0+16h .. k0+16h+15
        v16h a = *(const v16h*)(aRow + k0 + 16 * h);

        // B 32x16 f16 frag: lane holds contiguous K = k0+16h .. k0+16h+15
        v16h bfrag = *(const v16h*)(wRow + k0 + 16 * h);

        acc = __builtin_amdgcn_wmma_f32_16x16x32_f16(
                  false, a, false, bfrag, (short)0, acc, false, false);
    }

    const int nCol = nT * 16 + r;
    if (ACT == 0) {
        f16* out = (f16*)outv;
#pragma unroll
        for (int i = 0; i < 8; ++i) {
            int m = mT * 16 + 8 * h + i;               // D: M = vgpr + 8*h
            out[(size_t)m * N + nCol] = (f16)fmaxf(acc[i], 0.0f);
        }
    } else {
        float* out = (float*)outv;
#pragma unroll
        for (int i = 0; i < 8; ++i) {
            int m = mT * 16 + 8 * h + i;
            out[(size_t)m * N + nCol] = 1.0f / (1.0f + expf(-acc[i]));
        }
    }
}

extern "C" void kernel_launch(void* const* d_in, const int* in_sizes, int n_in,
                              void* d_out, int out_size, void* d_ws, size_t ws_size,
                              hipStream_t stream) {
    (void)in_sizes; (void)n_in; (void)out_size; (void)ws_size;

    const int*   stu_id = (const int*)  d_in[0];
    /* d_in[1] exer_id, d_in[2] kn_emb: unused by the reference */
    const int*   q_idx  = (const int*)  d_in[3];
    const float* emb    = (const float*)d_in[4];
    const float* fm     = (const float*)d_in[5];
    const float* w1     = (const float*)d_in[6];
    const float* b1     = (const float*)d_in[7];
    const float* w2     = (const float*)d_in[8];
    const float* b2     = (const float*)d_in[9];
    const float* w3     = (const float*)d_in[10];
    const float* b3     = (const float*)d_in[11];

    // Workspace carve-up (all regions fully overwritten every call)
    char* ws = (char*)d_ws;
    f16* C16 = (f16*)ws; ws += (size_t)BN * NOUT * sizeof(f16);   // 8 MB
    f16* h1  = (f16*)ws; ws += (size_t)BN * 256  * sizeof(f16);   // 2 MB
    f16* h2  = (f16*)ws; ws += (size_t)BN * 128  * sizeof(f16);   // 1 MB
    f16* w1h = (f16*)ws; ws += (size_t)256 * NOUT * sizeof(f16);
    f16* w2h = (f16*)ws; ws += (size_t)128 * 256  * sizeof(f16);
    f16* w3h = (f16*)ws; ws += (size_t)NOUT * 128 * sizeof(f16);

    // 1) weights -> f16
    cvt_f32_to_f16<<<(256 * NOUT + 255) / 256, 256, 0, stream>>>(w1, w1h, 256 * NOUT);
    cvt_f32_to_f16<<<(128 * 256  + 255) / 256, 256, 0, stream>>>(w2, w2h, 128 * 256);
    cvt_f32_to_f16<<<(NOUT * 128 + 255) / 256, 256, 0, stream>>>(w3, w3h, NOUT * 128);

    // 2) Choquet stage -> C (f16)
    choquet_kernel<<<dim3(NOUT / 256, BN), 256, 0, stream>>>(stu_id, emb, q_idx, fm, C16);

    // 3) MLP: three WMMA GEMMs
    // GEMM1: (4096x1024)x(1024x256) -> h1, relu.   tiles = 256*16 = 4096
    gemm_wmma<0><<<4096 / 8, 256, 0, stream>>>(C16, w1h, b1, h1, BN, 256, NOUT);
    // GEMM2: (4096x256)x(256x128) -> h2, relu.     tiles = 256*8  = 2048
    gemm_wmma<0><<<2048 / 8, 256, 0, stream>>>(h1, w2h, b2, h2, BN, 128, 256);
    // GEMM3: (4096x128)x(128x1024) -> out, sigmoid. tiles = 256*64 = 16384
    gemm_wmma<1><<<16384 / 8, 256, 0, stream>>>(h2, w3h, b3, d_out, BN, NOUT, 128);
}
